// ST_GNN_30382598652318
// MI455X (gfx1250) — matrix operand order repaired and checked by reading, not compiled, not run on hardware
//
#include <hip/hip_runtime.h>
#include <hip/hip_bf16.h>

#define B_  32
#define T_  12
#define N_  2048
#define D_  16
#define E_  2
#define H_  64
#define M_  12
#define HP_ 32
#define R_  (B_ * N_)          // 65536 flattened (b,n) rows

typedef __attribute__((ext_vector_type(16))) __bf16 v16bf;
typedef __attribute__((ext_vector_type(8)))  __bf16 v8bf;
typedef __attribute__((ext_vector_type(8)))  float  v8f;

union V16U { v16bf v; v8bf h[2]; };

__device__ __forceinline__ v8bf zero8() {
    union { unsigned long long u[2]; v8bf v; } z;
    z.u[0] = 0ull; z.u[1] = 0ull;
    return z.v;
}

// A operand (16Mx32K bf16), row-major source. Caller passes this lane's row base.
// lane<16: K {k0+0..7, k0+16..23}; lane>=16: K {k0+8..15, k0+24..31}  (ks = hiK*8)
__device__ __forceinline__ v16bf ldA(const __bf16* rowbase, int k0, int ks) {
    V16U u;
    u.h[0] = *(const v8bf*)(rowbase + k0 + ks);
    u.h[1] = *(const v8bf*)(rowbase + k0 + 16 + ks);
    return u.v;
}
// A operand with only K=0..15 valid (upper half zero-padded)
__device__ __forceinline__ v16bf ldA_k16(const __bf16* rowbase, int ks) {
    V16U u;
    u.h[0] = *(const v8bf*)(rowbase + ks);
    u.h[1] = zero8();
    return u.v;
}
// A operand, K=16 valid, converting from f32 source
__device__ __forceinline__ v16bf ldA_k16_f32(const float* rowbase, int ks) {
    V16U u;
#pragma unroll
    for (int j = 0; j < 8; ++j) u.h[0][j] = (__bf16)rowbase[ks + j];
    u.h[1] = zero8();
    return u.v;
}
// B operand (32Kx16N bf16), column-major source (K contiguous per column).
// lane<16: K {k0+0..7, k0+16..23}; lane>=16: K {k0+8..15, k0+24..31} (caller folds +8 via ks).
__device__ __forceinline__ v16bf ldB(const __bf16* colbase, int k0) {
    V16U u;
    u.h[0] = *(const v8bf*)(colbase + k0);
    u.h[1] = *(const v8bf*)(colbase + k0 + 16);
    return u.v;
}

__device__ __forceinline__ v8f wmma_bf16(v16bf a, v16bf b, v8f c) {
    return __builtin_amdgcn_wmma_f32_16x16x32_bf16(false, a, false, b, (short)0, c, false, false);
}

// ---------------- pack kernels ----------------

__global__ void pack_x_kernel(const float* __restrict__ x, __bf16* __restrict__ xT) {
    // xT[bt][d][m] = x[bt][m][d]   (column-major-K layout for WMMA B operand)
    const long total = (long)B_ * T_ * N_ * D_;
    for (long i = (long)blockIdx.x * blockDim.x + threadIdx.x; i < total;
         i += (long)gridDim.x * blockDim.x) {
        long d  = i & (D_ - 1);
        long m  = (i >> 4) & (N_ - 1);
        long bt = i >> 15;                       // N_*D_ == 1<<15
        xT[(bt * D_ + d) * N_ + m] = (__bf16)x[i];
    }
}

__global__ void pack_hops_kernel(const float* __restrict__ hops, __bf16* __restrict__ hopsb) {
    const long total = (long)E_ * N_ * N_;
    for (long i = (long)blockIdx.x * blockDim.x + threadIdx.x; i < total;
         i += (long)gridDim.x * blockDim.x)
        hopsb[i] = (__bf16)hops[i];
}

__global__ void pack_w_kernel(const float* __restrict__ Ws, const float* __restrict__ Wt,
                              const float* __restrict__ Wf, const float* __restrict__ Wz,
                              __bf16* __restrict__ WHS, __bf16* __restrict__ Wtb,
                              __bf16* __restrict__ WfX, __bf16* __restrict__ WfY,
                              __bf16* __restrict__ Wzb) {
    const int tid = threadIdx.x;
    // WHS[e][h][k32] = sum_hh Wf[h][16+hh] * Ws[e][hh][k]  (k<16; zero pad k>=16)
    for (int i = tid; i < E_ * H_ * 32; i += blockDim.x) {
        int k = i & 31, h = (i >> 5) & (H_ - 1), e = i >> 11;
        float s = 0.f;
        if (k < D_)
            for (int hh = 0; hh < H_; ++hh)
                s += Wf[h * (D_ + 2 * H_) + D_ + hh] * Ws[(e * H_ + hh) * D_ + k];
        WHS[i] = (__bf16)s;
    }
    for (int i = tid; i < M_ * H_ * H_; i += blockDim.x) Wtb[i] = (__bf16)Wt[i];
    for (int i = tid; i < H_ * 32; i += blockDim.x) {
        int k = i & 31, h = i >> 5;
        WfX[i] = (__bf16)(k < D_ ? Wf[h * (D_ + 2 * H_) + k] : 0.f);
    }
    for (int i = tid; i < H_ * H_; i += blockDim.x) {
        int kk = i & (H_ - 1), h = i >> 6;
        WfY[i] = (__bf16)Wf[h * (D_ + 2 * H_) + D_ + H_ + kk];
    }
    for (int i = tid; i < H_ * H_ * M_; i += blockDim.x) Wzb[i] = (__bf16)Wz[i];
}

// ---------------- per-timestep fused kernel ----------------
// Each wave owns 32 rows (2 sub-tiles of 16) so the x/B operand of the dominant
// aggregation GEMM is reused for 2 WMMAs per load (3 b128 loads per WMMA).
// Yst[t] = relu( X_t@WfX.T + sum_e agg_e@WHS[e].T + relu(sum_k Yprev[t-k]@Wt[k-1].T)@WfY.T + bf )

__global__ __launch_bounds__(256) void st_step_kernel(
    const float*  __restrict__ x,      // (B,T,N,D) f32
    const __bf16* __restrict__ hopsb,  // (E,N,N)
    const __bf16* __restrict__ xT,     // (B,T,D,N)
    const __bf16* __restrict__ WHS,    // (E,H,32)
    const __bf16* __restrict__ Wtb,    // (M,H,H)
    const __bf16* __restrict__ WfX,    // (H,32)
    const __bf16* __restrict__ WfY,    // (H,H)
    const float*  __restrict__ bfv,    // (H)
    __bf16*       __restrict__ Yprev,  // (M,R,H)
    int t)
{
    __shared__ __align__(16) __bf16 s_agg[8][16 * 16];
    __shared__ __align__(16) __bf16 s_y[8][16 * 64];

    const int lane = threadIdx.x & 31;
    const int w    = threadIdx.x >> 5;
    const int tile = blockIdx.x * 8 + w;   // 32-row tile
    const int row0 = tile * 32;
    const int b    = row0 >> 11;           // row0 / N_ ; 32 | N_ so tile never crosses b
    const int n0   = row0 & (N_ - 1);
    const int rl   = lane & 15;
    const int hiK  = (lane >> 4) & 1;
    const int ks   = hiK * 8;
    const int k16  = hiK * 16;
    const int hi8  = hiK * 8;

    // output accumulators: 2 sub-tiles x 4 col tiles, init with bias
    v8f acc[2][4];
#pragma unroll
    for (int c = 0; c < 4; ++c) {
        float bv = bfv[c * 16 + rl];
#pragma unroll
        for (int s = 0; s < 2; ++s)
#pragma unroll
            for (int r = 0; r < 8; ++r) acc[s][c][r] = bv;
    }

    // ---- graph aggregation: agg_e = hops[e, n0:n0+32, :] @ x[b,t]; acc += agg_e @ WHS[e].T
#pragma unroll
    for (int e = 0; e < E_; ++e) {
        v8f agg[2];
#pragma unroll
        for (int s = 0; s < 2; ++s)
#pragma unroll
            for (int r = 0; r < 8; ++r) agg[s][r] = 0.f;
        const __bf16* Arow0 = hopsb + ((size_t)e * N_ + (size_t)(n0 + rl)) * N_;
        const __bf16* Arow1 = Arow0 + (size_t)16 * N_;
        const __bf16* Bcol  = xT + (((size_t)b * T_ + t) * D_ + rl) * (size_t)N_ + ks;
        for (int k0 = 0; k0 < N_; k0 += 32) {
            __builtin_prefetch(Arow0 + k0 + 512, 0, 1);   // speculative; safe if OOB
            __builtin_prefetch(Arow1 + k0 + 512, 0, 1);
            v16bf bb = ldB(Bcol, k0);                     // shared by both sub-tiles
            agg[0] = wmma_bf16(ldA(Arow0, k0, ks), bb, agg[0]);
            agg[1] = wmma_bf16(ldA(Arow1, k0, ks), bb, agg[1]);
        }
#pragma unroll
        for (int s = 0; s < 2; ++s) {
            // transpose agg tile (lane = d-col) into row-major LDS (wave-private, in-order)
#pragma unroll
            for (int r = 0; r < 8; ++r)
                s_agg[w][(r + hi8) * 16 + rl] = (__bf16)agg[s][r];
            v16bf a2 = ldA_k16(&s_agg[w][rl * 16], ks);
#pragma unroll
            for (int c = 0; c < 4; ++c) {
                const __bf16* Bc = WHS + ((size_t)e * H_ + (size_t)(c * 16 + rl)) * 32 + ks;
                acc[s][c] = wmma_bf16(a2, ldB(Bc, 0), acc[s][c]);
            }
        }
    }

    // ---- X_t @ WfX.T  (K=16 padded, convert f32 -> bf16 in registers)
#pragma unroll
    for (int s = 0; s < 2; ++s) {
        const float* xrow = x + (((size_t)b * T_ + t) * N_ + (size_t)(n0 + s * 16 + rl)) * D_;
        v16bf ax = ldA_k16_f32(xrow, ks);
#pragma unroll
        for (int c = 0; c < 4; ++c) {
            const __bf16* Bc = WfX + (size_t)(c * 16 + rl) * 32 + ks;
            acc[s][c] = wmma_bf16(ax, ldB(Bc, 0), acc[s][c]);
        }
    }

    // ---- temporal: Y_t = relu(sum_k Yprev[t-k] @ Wt[k-1].T);  acc += Y_t @ WfY.T
    if (t > 0) {
#pragma unroll
        for (int s = 0; s < 2; ++s) {
            v8f ya[4];
#pragma unroll
            for (int c = 0; c < 4; ++c)
#pragma unroll
                for (int r = 0; r < 8; ++r) ya[c][r] = 0.f;
            for (int k = 1; k <= t; ++k) {
                const __bf16* Arow =
                    Yprev + ((size_t)(t - k) * R_ + (size_t)(row0 + s * 16 + rl)) * H_;
                const __bf16* Wk = Wtb + (size_t)(k - 1) * H_ * H_;
#pragma unroll
                for (int c = 0; c < 4; ++c) {
                    const __bf16* Bc = Wk + (size_t)(c * 16 + rl) * H_ + ks;
                    ya[c] = wmma_bf16(ldA(Arow, 0,  ks), ldB(Bc, 0),  ya[c]);
                    ya[c] = wmma_bf16(ldA(Arow, 32, ks), ldB(Bc, 32), ya[c]);
                }
            }
            // relu -> row-major bf16 tile in LDS (A-operand staging)
#pragma unroll
            for (int c = 0; c < 4; ++c)
#pragma unroll
                for (int r = 0; r < 8; ++r) {
                    float v = ya[c][r];
                    s_y[w][(r + hi8) * 64 + c * 16 + rl] = (__bf16)(v > 0.f ? v : 0.f);
                }
            const __bf16* Ay = &s_y[w][rl * 64];
#pragma unroll
            for (int c = 0; c < 4; ++c) {
                const __bf16* Bc = WfY + (size_t)(c * 16 + rl) * H_ + ks;
                acc[s][c] = wmma_bf16(ldA(Ay, 0,  ks), ldB(Bc, 0),  acc[s][c]);
                acc[s][c] = wmma_bf16(ldA(Ay, 32, ks), ldB(Bc, 32), acc[s][c]);
            }
        }
    }

    // ---- Yst = relu(acc) -> Yprev[t] (bf16)
    __bf16* Yout = Yprev + (size_t)t * R_ * H_;
#pragma unroll
    for (int s = 0; s < 2; ++s)
#pragma unroll
        for (int c = 0; c < 4; ++c)
#pragma unroll
            for (int r = 0; r < 8; ++r) {
                float v = acc[s][c][r];
                Yout[(size_t)(row0 + s * 16 + r + hi8) * H_ + c * 16 + rl] =
                    (__bf16)(v > 0.f ? v : 0.f);
            }
}

// ---------------- Wz GEMM + MLP head ----------------

__global__ __launch_bounds__(256) void st_head_kernel(
    const __bf16* __restrict__ Yprev,  // (M,R,H)
    const __bf16* __restrict__ Wzb,    // (H, H*M)
    const float*  __restrict__ bz,
    const float*  __restrict__ Wp1, const float* __restrict__ bp1,
    const float*  __restrict__ Wp2, const float* __restrict__ bp2,
    float*        __restrict__ out)    // arrivals (R) then departures (R)
{
    __shared__ __align__(16) float s_fh[8][16 * 64];

    const int lane = threadIdx.x & 31;
    const int w    = threadIdx.x >> 5;
    const int tile = blockIdx.x * 8 + w;
    const int row0 = tile * 16;
    const int rl   = lane & 15;
    const int hiK  = (lane >> 4) & 1;
    const int ks   = hiK * 8;
    const int k16  = hiK * 16;
    const int hi8  = hiK * 8;

    v8f acc[4];
#pragma unroll
    for (int c = 0; c < 4; ++c) {
        float bv = bz[c * 16 + rl];
#pragma unroll
        for (int r = 0; r < 8; ++r) acc[c][r] = bv;
    }
    // Fh = relu(Y_cat @ Wz.T + bz), K = H*M = 768 walked per t-block
    for (int t = 0; t < M_; ++t) {
        const __bf16* Arow = Yprev + ((size_t)t * R_ + (size_t)(row0 + rl)) * H_;
#pragma unroll
        for (int c = 0; c < 4; ++c) {
            const __bf16* Bc = Wzb + (size_t)(c * 16 + rl) * (H_ * M_) + t * H_ + ks;
            acc[c] = wmma_bf16(ldA(Arow, 0,  ks), ldB(Bc, 0),  acc[c]);
            acc[c] = wmma_bf16(ldA(Arow, 32, ks), ldB(Bc, 32), acc[c]);
        }
    }
#pragma unroll
    for (int c = 0; c < 4; ++c)
#pragma unroll
        for (int r = 0; r < 8; ++r) {
            float v = acc[c][r];
            s_fh[w][(r + hi8) * 64 + c * 16 + rl] = (v > 0.f ? v : 0.f);
        }

    // head: 16 lanes, one row each (fp32 VALU; tiny)
    if (lane < 16) {
        const float* fh = &s_fh[w][lane * 64];
        float h1[HP_];
        for (int p = 0; p < HP_; ++p) {
            float s = bp1[p];
#pragma unroll 8
            for (int h = 0; h < H_; ++h) s += fh[h] * Wp1[p * H_ + h];
            h1[p] = s > 0.f ? s : 0.f;
        }
        float a0 = bp2[0], a1 = bp2[1];
#pragma unroll
        for (int p = 0; p < HP_; ++p) { a0 += h1[p] * Wp2[p]; a1 += h1[p] * Wp2[HP_ + p]; }
        int rr = row0 + lane;
        out[rr]      = a0;   // arrivals
        out[R_ + rr] = a1;   // departures
    }
}

// ---------------- launch ----------------

extern "C" void kernel_launch(void* const* d_in, const int* in_sizes, int n_in,
                              void* d_out, int out_size, void* d_ws, size_t ws_size,
                              hipStream_t stream) {
    const float* x    = (const float*)d_in[0];
    // d_in[1] = adjacency (unused by reference)
    const float* hops = (const float*)d_in[2];
    const float* Ws   = (const float*)d_in[3];
    const float* Wt   = (const float*)d_in[4];
    const float* Wf   = (const float*)d_in[5];
    const float* bfv  = (const float*)d_in[6];
    const float* Wz   = (const float*)d_in[7];
    const float* bz   = (const float*)d_in[8];
    const float* Wp1  = (const float*)d_in[9];
    const float* bp1  = (const float*)d_in[10];
    const float* Wp2  = (const float*)d_in[11];
    const float* bp2  = (const float*)d_in[12];
    float* out = (float*)d_out;

    char* ws = (char*)d_ws;
    size_t off = 0;
    __bf16* xT    = (__bf16*)(ws + off); off += (size_t)B_ * T_ * D_ * N_ * 2;  // 25.2 MB
    __bf16* hopsb = (__bf16*)(ws + off); off += (size_t)E_ * N_ * N_ * 2;       // 16.8 MB
    __bf16* Yprev = (__bf16*)(ws + off); off += (size_t)M_ * R_ * H_ * 2;       // 100.7 MB
    __bf16* WHS   = (__bf16*)(ws + off); off += (size_t)E_ * H_ * 32 * 2;
    __bf16* Wtb   = (__bf16*)(ws + off); off += (size_t)M_ * H_ * H_ * 2;
    __bf16* WfX   = (__bf16*)(ws + off); off += (size_t)H_ * 32 * 2;
    __bf16* WfY   = (__bf16*)(ws + off); off += (size_t)H_ * H_ * 2;
    __bf16* Wzb   = (__bf16*)(ws + off); off += (size_t)H_ * H_ * M_ * 2;

    pack_x_kernel<<<2048, 256, 0, stream>>>(x, xT);
    pack_hops_kernel<<<2048, 256, 0, stream>>>(hops, hopsb);
    pack_w_kernel<<<1, 256, 0, stream>>>(Ws, Wt, Wf, Wz, WHS, Wtb, WfX, WfY, Wzb);

    for (int t = 0; t < M_; ++t)
        st_step_kernel<<<R_ / 32 / 8, 256, 0, stream>>>(x, hopsb, xT, WHS, Wtb, WfX, WfY,
                                                        bfv, Yprev, t);

    st_head_kernel<<<R_ / 16 / 8, 256, 0, stream>>>(Yprev, Wzb, bz, Wp1, bp1, Wp2, bp2, out);
}
